// NSABlock_73375221285369
// MI455X (gfx1250) — matrix operand order repaired and checked by reading, not compiled, not run on hardware
//
#include <hip/hip_runtime.h>
#include <math.h>

// Problem constants (from reference)
#define H   16
#define DH  64
#define BSZ 4
#define WIN 32
#define B_  2
#define S_  1024
#define D_  1024
#define NB  (S_/BSZ)     // 256 blocks
#define NC  (NB+1)       // 257 compressed ctx (mem + blocks)
#define ROWS (B_*S_)     // 2048
#define SCALE 0.125f     // DH^-0.5

typedef __attribute__((ext_vector_type(16))) __bf16 bf16x16;
typedef __attribute__((ext_vector_type(8)))  __bf16 bf16x8;
typedef __attribute__((ext_vector_type(8)))  float  floatx8;

// ---------------------------------------------------------------------------
// Weight convert+transpose:  Wt[n*K + k] = bf16(W[k*N + n])
// ---------------------------------------------------------------------------
__global__ void convT_kernel(const float* __restrict__ W, __bf16* __restrict__ Wt,
                             int K, int N) {
  const size_t idx = (size_t)blockIdx.x * blockDim.x + threadIdx.x;
  if (idx >= (size_t)K * N) return;
  const int k = (int)(idx / N);
  const int n = (int)(idx % N);
  Wt[(size_t)n * K + k] = (__bf16)W[idx];
}

// ---------------------------------------------------------------------------
// Fused lincomb + RMSNorm.  Writes optional f32 xr and bf16 h (GEMM A operand).
// One block (256 thr) per row.
// ---------------------------------------------------------------------------
__global__ __launch_bounds__(256)
void rms_kernel(const float* __restrict__ xa, const float* __restrict__ xb,
                const float* __restrict__ lam,
                float* __restrict__ xr_out, __bf16* __restrict__ hb_out) {
  __shared__ float red[8];
  __shared__ float rms_s;
  const int row = blockIdx.x;
  float l0 = 1.0f, l1 = 0.0f;
  if (lam) { l0 = lam[0]; l1 = lam[1]; }
  const size_t base = (size_t)row * D_ + threadIdx.x * 4;
  float4 va = *reinterpret_cast<const float4*>(&xa[base]);
  float4 vb = make_float4(0.f, 0.f, 0.f, 0.f);
  if (xb) vb = *reinterpret_cast<const float4*>(&xb[base]);
  float v0 = l0 * va.x + l1 * vb.x;
  float v1 = l0 * va.y + l1 * vb.y;
  float v2 = l0 * va.z + l1 * vb.z;
  float v3 = l0 * va.w + l1 * vb.w;
  float ss = v0 * v0 + v1 * v1 + v2 * v2 + v3 * v3;
  #pragma unroll
  for (int off = 16; off > 0; off >>= 1) ss += __shfl_down(ss, off, 32);
  if ((threadIdx.x & 31) == 0) red[threadIdx.x >> 5] = ss;
  __syncthreads();
  if (threadIdx.x == 0) {
    float t = 0.f;
    #pragma unroll
    for (int i = 0; i < 8; ++i) t += red[i];
    rms_s = rsqrtf(t / (float)D_ + 1e-6f);
  }
  __syncthreads();
  const float r = rms_s;
  if (xr_out) {
    float4 o = make_float4(v0, v1, v2, v3);
    *reinterpret_cast<float4*>(&xr_out[base]) = o;
  }
  hb_out[base + 0] = (__bf16)(v0 * r);
  hb_out[base + 1] = (__bf16)(v1 * r);
  hb_out[base + 2] = (__bf16)(v2 * r);
  hb_out[base + 3] = (__bf16)(v3 * r);
}

// epilogue helper
__device__ __forceinline__ float apply_epi(float v, int epi, const float* resid,
                                           size_t cidx) {
  if (epi == 1)      v = 1.0f / (1.0f + __expf(-v));
  else if (epi == 2) { v = v > 0.0f ? v : 0.0f; v = v * v; }
  else if (epi == 3) v += resid[cidx];
  return v;
}

// ---------------------------------------------------------------------------
// Main bf16 WMMA GEMM, N-tile = 64.
//   C(MxN) = epi( A(MxK,bf16) @ Wt(NxK,bf16)^T [+resid] )
// 4 waves/block; each wave: 16 rows x 64 cols (4 wmma / 32-K step).
// bf16 B tile (64 n x 32 k, n-major) streamed global->LDS with
// GLOBAL_LOAD_ASYNC_TO_LDS_B128, double buffered via s_wait_asynccnt.
// A fragment: two contiguous 16B bf16 loads (no conversions anywhere).
// Output to f32 C or bf16 Cb.   M%64==0, N%64==0, K%32==0.
// ---------------------------------------------------------------------------
__global__ __launch_bounds__(128)
void gemm_bf16_wmma_n64(const __bf16* __restrict__ A, const __bf16* __restrict__ Wt,
                        float* __restrict__ C, __bf16* __restrict__ Cb,
                        int M, int N, int K,
                        const float* __restrict__ resid, int epi) {
  __shared__ __align__(32) __bf16 Bs[2][64 * 32];
  const int lane = threadIdx.x & 31;
  const int wave = threadIdx.x >> 5;
  const int hi   = lane >> 4;
  const int lo   = lane & 15;
  const int n0   = blockIdx.x * 64;
  const int m0   = blockIdx.y * 64 + wave * 16;
  (void)M;

  floatx8 acc0 = {}, acc1 = {}, acc2 = {}, acc3 = {};
  const int ksteps = K >> 5;
  const __bf16* arow = &A[(size_t)(m0 + lo) * K];
  const unsigned lds0 = (unsigned)(size_t)(&Bs[0][0]);
  const unsigned lds1 = (unsigned)(size_t)(&Bs[1][0]);

  // async-issue one 64x32 bf16 tile (256 x 16B chunks; 2 per thread)
  auto issue_tile = [&](int kb, unsigned ldsbase) {
    const int kbase = kb << 5;
    #pragma unroll
    for (int i = 0; i < 2; ++i) {
      const int c  = threadIdx.x + 128 * i;   // 0..255
      const int n  = c >> 2;                  // 0..63
      const int k8 = (c & 3) << 3;            // 0,8,16,24
      const __bf16* gp = &Wt[(size_t)(n0 + n) * K + kbase + k8];
      const unsigned lp = ldsbase + ((unsigned)c << 4);
      asm volatile("global_load_async_to_lds_b128 %0, %1, off"
                   :: "v"(lp), "v"(gp) : "memory");
    }
  };

  issue_tile(0, lds0);
  for (int kb = 0; kb < ksteps; ++kb) {
    const int buf = kb & 1;
    if (kb + 1 < ksteps) {
      issue_tile(kb + 1, buf ? lds0 : lds1);
      asm volatile("s_wait_asynccnt 0x2" ::: "memory");  // previous tile done
    } else {
      asm volatile("s_wait_asynccnt 0x0" ::: "memory");
    }
    __syncthreads();

    // A fragment (contiguous k = 16*hi + j, same order as B):
    // hi=0: K={0..15}; hi=1: K={16..31} -> two contiguous 8-elt runs
    const int kbase = kb << 5;
    bf16x8 alo = *reinterpret_cast<const bf16x8*>(arow + kbase + 16 * hi);
    bf16x8 ahi = *reinterpret_cast<const bf16x8*>(arow + kbase + 16 * hi + 8);
    if (kb + 1 < ksteps) __builtin_prefetch(arow + kbase + 32, 0, 0);
    bf16x16 af;
    #pragma unroll
    for (int j = 0; j < 8; ++j) { af[j] = alo[j]; af[j + 8] = ahi[j]; }

    // B fragments: elem j -> Klocal = 16*hi + j at fixed Nlocal = t*16+lo
    // n-major LDS tile => 16 contiguous bf16 per fragment
    const __bf16* bsb = &Bs[buf][0];
    bf16x16 bf0 = *reinterpret_cast<const bf16x16*>(bsb + (0  + lo) * 32 + 16 * hi);
    bf16x16 bf1 = *reinterpret_cast<const bf16x16*>(bsb + (16 + lo) * 32 + 16 * hi);
    bf16x16 bf2 = *reinterpret_cast<const bf16x16*>(bsb + (32 + lo) * 32 + 16 * hi);
    bf16x16 bf3 = *reinterpret_cast<const bf16x16*>(bsb + (48 + lo) * 32 + 16 * hi);

    acc0 = __builtin_amdgcn_wmma_f32_16x16x32_bf16(false, af, false, bf0, (short)0, acc0, false, false);
    acc1 = __builtin_amdgcn_wmma_f32_16x16x32_bf16(false, af, false, bf1, (short)0, acc1, false, false);
    acc2 = __builtin_amdgcn_wmma_f32_16x16x32_bf16(false, af, false, bf2, (short)0, acc2, false, false);
    acc3 = __builtin_amdgcn_wmma_f32_16x16x32_bf16(false, af, false, bf3, (short)0, acc3, false, false);
    __syncthreads();
  }

  // store: VGPR r -> M = m0 + r + 8*hi, N = n0 + t*16 + lo
  #pragma unroll
  for (int r = 0; r < 8; ++r) {
    const int m = m0 + r + 8 * hi;
    float vv;
    size_t ci;
    ci = (size_t)m * N + n0 + lo;
    vv = apply_epi(acc0[r], epi, resid, ci);
    if (Cb) Cb[ci] = (__bf16)vv; else C[ci] = vv;
    ci = (size_t)m * N + n0 + 16 + lo;
    vv = apply_epi(acc1[r], epi, resid, ci);
    if (Cb) Cb[ci] = (__bf16)vv; else C[ci] = vv;
    ci = (size_t)m * N + n0 + 32 + lo;
    vv = apply_epi(acc2[r], epi, resid, ci);
    if (Cb) Cb[ci] = (__bf16)vv; else C[ci] = vv;
    ci = (size_t)m * N + n0 + 48 + lo;
    vv = apply_epi(acc3[r], epi, resid, ci);
    if (Cb) Cb[ci] = (__bf16)vv; else C[ci] = vv;
  }
}

// ---------------------------------------------------------------------------
// Narrow-N WMMA GEMM (N%16==0) for the gating projection (N=48), bf16 inputs.
// ---------------------------------------------------------------------------
__global__ __launch_bounds__(128)
void gemm_bf16_wmma_n16(const __bf16* __restrict__ A, const __bf16* __restrict__ Wt,
                        float* __restrict__ C, int M, int N, int K, int epi) {
  __shared__ __align__(32) __bf16 Bs[16 * 32];
  const int lane = threadIdx.x & 31;
  const int wave = threadIdx.x >> 5;
  const int hi   = lane >> 4;
  const int lo   = lane & 15;
  const int n0   = blockIdx.x * 16;
  const int m0   = blockIdx.y * 64 + wave * 16;
  (void)M;

  floatx8 acc = {};
  const int ksteps = K >> 5;
  const __bf16* arow = &A[(size_t)(m0 + lo) * K];

  for (int kb = 0; kb < ksteps; ++kb) {
    const int kbase = kb << 5;
    if (threadIdx.x < 64) {   // stage 16x32 bf16 tile (64 x 16B chunks)
      const int c  = threadIdx.x;
      const int n  = c >> 2;
      const int k8 = (c & 3) << 3;
      *reinterpret_cast<bf16x8*>(&Bs[n * 32 + k8]) =
          *reinterpret_cast<const bf16x8*>(&Wt[(size_t)(n0 + n) * K + kbase + k8]);
    }
    __syncthreads();

    bf16x8 alo = *reinterpret_cast<const bf16x8*>(arow + kbase + 16 * hi);
    bf16x8 ahi = *reinterpret_cast<const bf16x8*>(arow + kbase + 16 * hi + 8);
    bf16x16 af, bfv;
    #pragma unroll
    for (int j = 0; j < 8; ++j) { af[j] = alo[j]; af[j + 8] = ahi[j]; }
    bfv = *reinterpret_cast<const bf16x16*>(&Bs[lo * 32 + 16 * hi]);

    acc = __builtin_amdgcn_wmma_f32_16x16x32_bf16(false, af, false, bfv,
                                                  (short)0, acc, false, false);
    __syncthreads();
  }

  #pragma unroll
  for (int r = 0; r < 8; ++r) {
    const int m = m0 + r + 8 * hi;
    const size_t ci = (size_t)m * N + n0 + lo;
    C[ci] = apply_epi(acc[r], epi, nullptr, ci);
  }
}

// ---------------------------------------------------------------------------
// RoPE: q,k (B,S,H,DH) -> qr,kr.  One thread per (b,s,h,d<32).
// ---------------------------------------------------------------------------
__global__ void rope_kernel(const float* __restrict__ q, const float* __restrict__ k,
                            float* __restrict__ qr, float* __restrict__ kr) {
  const size_t idx = (size_t)blockIdx.x * blockDim.x + threadIdx.x;
  if (idx >= (size_t)B_ * S_ * H * 32) return;
  const int dd = idx & 31;
  const size_t t = idx >> 5;            // (b*S+s)*H + h
  const size_t bs = t >> 4;             // b*S + s
  const int s = (int)(bs & (S_ - 1));
  const float inv = __expf(-(float)dd * (logf(10000.0f) / 32.0f));
  float sn, cs;
  __sincosf((float)s * inv, &sn, &cs);
  const size_t base = t * DH + dd;
  float x1 = q[base], x2 = q[base + 32];
  qr[base] = x1 * cs - x2 * sn;
  qr[base + 32] = x1 * sn + x2 * cs;
  x1 = k[base]; x2 = k[base + 32];
  kr[base] = x1 * cs - x2 * sn;
  kr[base + 32] = x1 * sn + x2 * cs;
}

// ---------------------------------------------------------------------------
// Block compression: kc/vc ((b,n,h,dout), n=0 is mem slot).
// ---------------------------------------------------------------------------
__global__ __launch_bounds__(128)
void compress_kernel(const float* __restrict__ k, const float* __restrict__ v,
                     const float* __restrict__ pos_k, const float* __restrict__ pos_v,
                     const float* __restrict__ Wck, const float* __restrict__ Wcv,
                     const float* __restrict__ mem_k, const float* __restrict__ mem_v,
                     float* __restrict__ kcf, float* __restrict__ vcf) {
  const size_t idx = (size_t)blockIdx.x * blockDim.x + threadIdx.x;
  if (idx >= (size_t)B_ * NC * H * DH) return;
  const int dout = idx & 63;
  size_t t = idx >> 6;
  const int h = (int)(t & 15); t >>= 4;
  const int n = (int)(t % NC);
  const int b = (int)(t / NC);
  if (n == 0) {
    kcf[idx] = mem_k[h * DH + dout];
    vcf[idx] = mem_v[h * DH + dout];
    return;
  }
  const int blk = n - 1;
  float ak = 0.f, av = 0.f;
  for (int tt = 0; tt < BSZ; ++tt) {
    const size_t kb = ((size_t)(b * S_ + blk * BSZ + tt) * H + h) * DH;
    const size_t pb = ((size_t)h * BSZ + tt) * DH;
    #pragma unroll 8
    for (int d = 0; d < DH; ++d) {
      const size_t wr = (size_t)(tt * DH + d) * DH + dout;
      ak += (k[kb + d] + pos_k[pb + d]) * Wck[wr];
      av += (v[kb + d] + pos_v[pb + d]) * Wcv[wr];
    }
  }
  kcf[idx] = ak;
  vcf[idx] = av;
}

// ---------------------------------------------------------------------------
// Compressed attention + top-4 selection.  One thread per (b,h,s).
// sel row: [i0,i1,i2,i3,own,validmask,_,_]
// ---------------------------------------------------------------------------
__global__ __launch_bounds__(128)
void cattn_kernel(const float* __restrict__ q, const float* __restrict__ kcf,
                  const float* __restrict__ vcf, float* __restrict__ cout,
                  int* __restrict__ sel) {
  const int idx = blockIdx.x * blockDim.x + threadIdx.x;
  if (idx >= B_ * H * S_) return;
  const int s = idx & (S_ - 1);
  const int h = (idx >> 10) & 15;
  const int b = idx >> 14;
  const size_t qbase = ((size_t)(b * S_ + s) * H + h) * DH;

  float qv[DH];
  #pragma unroll
  for (int d = 0; d < DH; ++d) qv[d] = q[qbase + d];

  const int nvis = (s >= BSZ - 1) ? ((s - (BSZ - 1)) >> 2) + 1 : 0;
  float mx = -INFINITY;
  float tv0 = -INFINITY, tv1 = -INFINITY, tv2 = -INFINITY, tv3 = -INFINITY;
  int ti0 = 0, ti1 = 1, ti2 = 2, ti3 = 3;

  for (int n = 0; n <= nvis; ++n) {
    const size_t kb = ((size_t)(b * NC + n) * H + h) * DH;
    float acc = 0.f;
    #pragma unroll
    for (int d = 0; d < DH; ++d) acc += qv[d] * kcf[kb + d];
    acc *= SCALE;
    mx = fmaxf(mx, acc);
    if (n >= 1) {
      const int i = n - 1;
      if (acc > tv0)      { tv3=tv2; ti3=ti2; tv2=tv1; ti2=ti1; tv1=tv0; ti1=ti0; tv0=acc; ti0=i; }
      else if (acc > tv1) { tv3=tv2; ti3=ti2; tv2=tv1; ti2=ti1; tv1=acc; ti1=i; }
      else if (acc > tv2) { tv3=tv2; ti3=ti2; tv2=acc; ti2=i; }
      else if (acc > tv3) { tv3=acc; ti3=i; }
    }
  }

  float oacc[DH];
  #pragma unroll
  for (int d = 0; d < DH; ++d) oacc[d] = 0.f;
  float sum = 0.f;
  for (int n = 0; n <= nvis; ++n) {
    const size_t kb = ((size_t)(b * NC + n) * H + h) * DH;
    float acc = 0.f;
    #pragma unroll
    for (int d = 0; d < DH; ++d) acc += qv[d] * kcf[kb + d];
    const float e = __expf(acc * SCALE - mx);
    sum += e;
    #pragma unroll
    for (int d = 0; d < DH; ++d) oacc[d] += e * vcf[kb + d];
  }
  const float inv = 1.0f / sum;
  #pragma unroll
  for (int d = 0; d < DH; ++d) cout[qbase + d] = oacc[d] * inv;

  int* sp = sel + (size_t)idx * 8;
  sp[0] = ti0; sp[1] = ti1; sp[2] = ti2; sp[3] = ti3;
  sp[4] = s >> 2;
  sp[5] = (tv0 > -1e30f ? 1 : 0) | (tv1 > -1e30f ? 2 : 0) |
          (tv2 > -1e30f ? 4 : 0) | (tv3 > -1e30f ? 8 : 0) | 16;
}

// ---------------------------------------------------------------------------
// Fine (selected-block) attention.  One thread per (b,h,s).
// ---------------------------------------------------------------------------
__global__ __launch_bounds__(128)
void fattn_kernel(const float* __restrict__ qr, const float* __restrict__ kr,
                  const float* __restrict__ v, const int* __restrict__ sel,
                  float* __restrict__ fout) {
  const int idx = blockIdx.x * blockDim.x + threadIdx.x;
  if (idx >= B_ * H * S_) return;
  const int s = idx & (S_ - 1);
  const int h = (idx >> 10) & 15;
  const int b = idx >> 14;
  const size_t qbase = ((size_t)(b * S_ + s) * H + h) * DH;

  float qv[DH];
  #pragma unroll
  for (int d = 0; d < DH; ++d) qv[d] = qr[qbase + d];

  const int* sp = sel + (size_t)idx * 8;
  int blks[5] = {sp[0], sp[1], sp[2], sp[3], sp[4]};
  const int vm = sp[5];

  float sc[20];
  float mx = -INFINITY;
  #pragma unroll
  for (int mi = 0; mi < 5; ++mi) {
    const int blk = blks[mi];
    const bool ok = (vm >> mi) & 1;
    #pragma unroll
    for (int tt = 0; tt < BSZ; ++tt) {
      const int kpos = blk * BSZ + tt;
      float a = -INFINITY;
      if (ok && kpos <= s) {
        const size_t kb = ((size_t)(b * S_ + kpos) * H + h) * DH;
        float acc = 0.f;
        #pragma unroll
        for (int d = 0; d < DH; ++d) acc += qv[d] * kr[kb + d];
        a = acc * SCALE;
      }
      sc[mi * BSZ + tt] = a;
      mx = fmaxf(mx, a);
    }
  }

  float oacc[DH];
  #pragma unroll
  for (int d = 0; d < DH; ++d) oacc[d] = 0.f;
  float sum = 0.f;
  #pragma unroll
  for (int mi = 0; mi < 5; ++mi) {
    const int blk = blks[mi];
    #pragma unroll
    for (int tt = 0; tt < BSZ; ++tt) {
      const float p = __expf(sc[mi * BSZ + tt] - mx);
      sum += p;
      if (p > 0.f) {
        const int kpos = blk * BSZ + tt;
        const size_t vb = ((size_t)(b * S_ + kpos) * H + h) * DH;
        #pragma unroll
        for (int d = 0; d < DH; ++d) oacc[d] += p * v[vb + d];
      }
    }
  }
  const float inv = 1.0f / sum;
  #pragma unroll
  for (int d = 0; d < DH; ++d) fout[qbase + d] = oacc[d] * inv;
}

// ---------------------------------------------------------------------------
// Sliding-window attention (WIN=32).  One thread per (b,h,s).
// ---------------------------------------------------------------------------
__global__ __launch_bounds__(128)
void sattn_kernel(const float* __restrict__ qr, const float* __restrict__ kr,
                  const float* __restrict__ v, float* __restrict__ sout) {
  const int idx = blockIdx.x * blockDim.x + threadIdx.x;
  if (idx >= B_ * H * S_) return;
  const int s = idx & (S_ - 1);
  const int h = (idx >> 10) & 15;
  const int b = idx >> 14;
  const size_t qbase = ((size_t)(b * S_ + s) * H + h) * DH;

  float qv[DH];
  #pragma unroll
  for (int d = 0; d < DH; ++d) qv[d] = qr[qbase + d];

  float sc[WIN];
  float mx = -INFINITY;
  #pragma unroll 4
  for (int w = 0; w < WIN; ++w) {
    const int sp = s - w;
    float a = -INFINITY;
    if (sp >= 0) {
      const size_t kb = ((size_t)(b * S_ + sp) * H + h) * DH;
      float acc = 0.f;
      #pragma unroll
      for (int d = 0; d < DH; ++d) acc += qv[d] * kr[kb + d];
      a = acc * SCALE;
    }
    sc[w] = a;
    mx = fmaxf(mx, a);
  }

  float oacc[DH];
  #pragma unroll
  for (int d = 0; d < DH; ++d) oacc[d] = 0.f;
  float sum = 0.f;
  #pragma unroll 4
  for (int w = 0; w < WIN; ++w) {
    const float p = __expf(sc[w] - mx);
    sum += p;
    if (p > 0.f) {
      const int sp = s - w;
      const size_t vb = ((size_t)(b * S_ + sp) * H + h) * DH;
      #pragma unroll
      for (int d = 0; d < DH; ++d) oacc[d] += p * v[vb + d];
    }
  }
  const float inv = 1.0f / sum;
  #pragma unroll
  for (int d = 0; d < DH; ++d) sout[qbase + d] = oacc[d] * inv;
}

// ---------------------------------------------------------------------------
// Gated combine: o = g0*cout + g1*fout + g2*sout -> bf16 (GEMM A operand)
// ---------------------------------------------------------------------------
__global__ void combine_kernel(const float* __restrict__ g,
                               const float* __restrict__ cout,
                               const float* __restrict__ fout,
                               const float* __restrict__ sout,
                               __bf16* __restrict__ ob) {
  const size_t idx = (size_t)blockIdx.x * blockDim.x + threadIdx.x;
  if (idx >= (size_t)ROWS * H * DH) return;
  const size_t t = idx >> 6;
  const int h = (int)(t & 15);
  const size_t bs = t >> 4;
  const size_t gb = bs * (3 * H) + h * 3;
  ob[idx] = (__bf16)(g[gb] * cout[idx] + g[gb + 1] * fout[idx] + g[gb + 2] * sout[idx]);
}

// ---------------------------------------------------------------------------
extern "C" void kernel_launch(void* const* d_in, const int* in_sizes, int n_in,
                              void* d_out, int out_size, void* d_ws, size_t ws_size,
                              hipStream_t stream) {
  (void)in_sizes; (void)n_in; (void)out_size; (void)ws_size;
  const float* x    = (const float*)d_in[0];
  const float* x0   = (const float*)d_in[2];
  const float* lam  = (const float*)d_in[3];
  const float* Wq   = (const float*)d_in[4];
  const float* Wk   = (const float*)d_in[5];
  const float* Wv   = (const float*)d_in[6];
  const float* posk = (const float*)d_in[7];
  const float* posv = (const float*)d_in[8];
  const float* Wck  = (const float*)d_in[9];
  const float* Wcv  = (const float*)d_in[10];
  const float* memk = (const float*)d_in[11];
  const float* memv = (const float*)d_in[12];
  const float* Wg   = (const float*)d_in[13];
  const float* Wo   = (const float*)d_in[14];
  const float* Wfc  = (const float*)d_in[15];
  const float* Wpr  = (const float*)d_in[16];
  float* out = (float*)d_out;

  // ---- workspace carving (byte cursor; every size is 256B-multiple) ----
  char* cur = (char*)d_ws;
  const size_t T = (size_t)ROWS * D_;   // 2M elements
  auto takeF = [&](size_t n) { float* p = (float*)cur;  cur += n * 4; return p; };
  auto takeB = [&](size_t n) { __bf16* p = (__bf16*)cur; cur += n * 2; return p; };
  auto takeI = [&](size_t n) { int* p = (int*)cur;      cur += n * 4; return p; };

  float*  xr   = takeF(T);
  float*  q    = takeF(T);
  float*  k    = takeF(T);
  float*  v    = takeF(T);
  float*  qr   = takeF(T);
  float*  kr   = takeF(T);
  float*  kcf  = takeF((size_t)B_ * NC * H * DH);
  float*  vcf  = takeF((size_t)B_ * NC * H * DH);
  float*  cout = takeF(T);
  float*  fout = takeF(T);
  float*  sout = takeF(T);
  float*  gbuf = takeF((size_t)ROWS * 3 * H);
  float*  x1   = takeF(T);
  int*    sel  = takeI((size_t)B_ * H * S_ * 8);
  __bf16* hb   = takeB(T);
  __bf16* ob   = takeB(T);
  __bf16* mb   = takeB(T);
  __bf16* actb = takeB(4 * T);
  __bf16* wqT  = takeB((size_t)D_ * D_);
  __bf16* wkT  = takeB((size_t)D_ * D_);
  __bf16* wvT  = takeB((size_t)D_ * D_);
  __bf16* wgT  = takeB((size_t)D_ * 3 * H);
  __bf16* woT  = takeB((size_t)D_ * D_);
  __bf16* wfcT = takeB((size_t)D_ * 4 * D_);
  __bf16* wprT = takeB((size_t)4 * D_ * D_);

  // ---- one-pass weight convert+transpose to bf16 [N][K] ----
  {
    const size_t n1 = (size_t)D_ * D_;
    const unsigned g1 = (unsigned)((n1 + 255) / 256);
    convT_kernel<<<g1, 256, 0, stream>>>(Wq, wqT, D_, D_);
    convT_kernel<<<g1, 256, 0, stream>>>(Wk, wkT, D_, D_);
    convT_kernel<<<g1, 256, 0, stream>>>(Wv, wvT, D_, D_);
    convT_kernel<<<g1, 256, 0, stream>>>(Wo, woT, D_, D_);
    const size_t ng = (size_t)D_ * 3 * H;
    convT_kernel<<<(unsigned)((ng + 255) / 256), 256, 0, stream>>>(Wg, wgT, D_, 3 * H);
    const size_t n4 = (size_t)D_ * 4 * D_;
    const unsigned g4 = (unsigned)((n4 + 255) / 256);
    convT_kernel<<<g4, 256, 0, stream>>>(Wfc, wfcT, D_, 4 * D_);
    convT_kernel<<<g4, 256, 0, stream>>>(Wpr, wprT, 4 * D_, D_);
  }

  rms_kernel<<<ROWS, 256, 0, stream>>>(x, x0, lam, xr, hb);

  dim3 blk(128);
  dim3 g64(D_ / 64, ROWS / 64);
  gemm_bf16_wmma_n64<<<g64, blk, 0, stream>>>(hb, wqT, q, nullptr, ROWS, D_, D_, nullptr, 0);
  gemm_bf16_wmma_n64<<<g64, blk, 0, stream>>>(hb, wkT, k, nullptr, ROWS, D_, D_, nullptr, 0);
  gemm_bf16_wmma_n64<<<g64, blk, 0, stream>>>(hb, wvT, v, nullptr, ROWS, D_, D_, nullptr, 0);

  {
    const size_t n = (size_t)B_ * S_ * H * 32;
    rope_kernel<<<(unsigned)((n + 255) / 256), 256, 0, stream>>>(q, k, qr, kr);
  }
  {
    const size_t n = (size_t)B_ * NC * H * DH;
    compress_kernel<<<(unsigned)((n + 127) / 128), 128, 0, stream>>>(
        k, v, posk, posv, Wck, Wcv, memk, memv, kcf, vcf);
  }

  cattn_kernel<<<(B_ * H * S_ + 127) / 128, 128, 0, stream>>>(q, kcf, vcf, cout, sel);
  fattn_kernel<<<(B_ * H * S_ + 127) / 128, 128, 0, stream>>>(qr, kr, v, sel, fout);
  sattn_kernel<<<(B_ * H * S_ + 127) / 128, 128, 0, stream>>>(qr, kr, v, sout);

  gemm_bf16_wmma_n16<<<dim3(3, ROWS / 64), blk, 0, stream>>>(hb, wgT, gbuf, ROWS, 3 * H, D_, 1);

  {
    const size_t n = (size_t)ROWS * H * DH;
    combine_kernel<<<(unsigned)((n + 255) / 256), 256, 0, stream>>>(gbuf, cout, fout, sout, ob);
  }

  gemm_bf16_wmma_n64<<<g64, blk, 0, stream>>>(ob, woT, x1, nullptr, ROWS, D_, D_, xr, 3);
  rms_kernel<<<ROWS, 256, 0, stream>>>(x1, nullptr, nullptr, nullptr, mb);
  gemm_bf16_wmma_n64<<<dim3(4 * D_ / 64, ROWS / 64), blk, 0, stream>>>(
      mb, wfcT, nullptr, actb, ROWS, 4 * D_, D_, nullptr, 2);
  gemm_bf16_wmma_n64<<<g64, blk, 0, stream>>>(actb, wprT, out, nullptr, ROWS, D_, 4 * D_, x1, 3);
}
